// VQEmbedding_16758962389518
// MI455X (gfx1250) — matrix operand order repaired and checked by reading, not compiled, not run on hardware
//
#include <hip/hip_runtime.h>
#include <hip/hip_bf16.h>

// ---------------------------------------------------------------------------
// VQ-VAE EMA codebook update, fused for MI455X (gfx1250, wave32, WMMA + TDM).
// Distances GEMM runs on v_wmma_f32_16x16x32_bf16; sampling is fused into the
// GEMM epilogue (streaming Gumbel-max) so the [N,K] distance matrix is never
// materialized. Codebook (B-matrix) tiles are staged into LDS by the Tensor
// Data Mover, double-buffered, so TDM DMA overlaps WMMA + epilogue work and
// the 8 waves of a workgroup share one copy of each B tile.
// dw = encodings^T @ x is a sparse scatter (5 atomics/row), not a dense GEMM.
// ---------------------------------------------------------------------------

#define KCODES 1024
#define DDIM   256
#define NROWS  32768   // B*T = 64*512
#define NSAMP  5
#define DECAYF 0.999f
#define EPSF   1e-5f
#define COMMITF 0.25f

typedef __attribute__((ext_vector_type(16))) __bf16 v16bf;
typedef __attribute__((ext_vector_type(8)))  float  v8f;
typedef __attribute__((ext_vector_type(4)))  unsigned int v4u;
typedef __attribute__((ext_vector_type(8)))  int  v8i_t;
typedef __attribute__((ext_vector_type(4)))  int  v4i_t;

#if __has_builtin(__builtin_amdgcn_tensor_load_to_lds) && \
    __has_builtin(__builtin_amdgcn_s_wait_tensorcnt)
#define VQ_USE_TDM 1
#else
#define VQ_USE_TDM 0
#endif

union FragBF16 {
    unsigned int u[8];
    uint4 q[2];
    v16bf v;
};

__device__ __forceinline__ unsigned short f32_to_bf16_rne(float f) {
    unsigned int u = __float_as_uint(f);
    unsigned int r = u + 0x7FFFu + ((u >> 16) & 1u);
    return (unsigned short)(r >> 16);
}

__device__ __forceinline__ unsigned int mix32(unsigned int h) {
    h ^= h >> 16; h *= 0x7feb352du;
    h ^= h >> 15; h *= 0x846ca68bu;
    h ^= h >> 16;
    return h;
}

// Monotonic float -> orderable u32, packed with the code index so a single
// u64 max is an argmax.
__device__ __forceinline__ unsigned long long pack_key(float f, unsigned int k) {
    unsigned int u = __float_as_uint(f);
    u = (u & 0x80000000u) ? ~u : (u | 0x80000000u);
    return (((unsigned long long)u) << 32) | (unsigned long long)k;
}

#if VQ_USE_TDM
// Issue a TDM load of one B tile: 16 codes x 256 dims of bf16 (8 KiB) from
// wb[ct*16 .. ct*16+15][0..255] into LDS at lds_off. 2-D descriptor per
// CDNA5 ISA 8.3/8.4: data_size=2B, tensor_dim0=256 (stride 256),
// tensor_dim1=1024, tile_dim0=256, tile_dim1=16.
__device__ __forceinline__ void tdm_load_b_tile(const unsigned short* wb,
                                                int ct, unsigned lds_off) {
    unsigned long long ga =
        (unsigned long long)(uintptr_t)(wb + (size_t)ct * 16 * DDIM);
    v4u g0;
    g0[0] = 1u;                                    // count=1, user D#
    g0[1] = lds_off;                               // lds_addr (bytes)
    g0[2] = (unsigned)(ga & 0xFFFFFFFFull);        // global_addr[31:0]
    g0[3] = (unsigned)((ga >> 32) & 0x01FFFFFFull) // global_addr[56:32]
          | (2u << 30);                            // type=2 ("image")
    v8i_t g1;
    g1[0] = 0x00010000;                            // wg_mask=0, data_size=1(2B)
    g1[1] = (int)((unsigned)DDIM << 16);           // tensor_dim0[15:0]=256
    g1[2] = (int)((unsigned)KCODES << 16);         // tensor_dim1[15:0]=1024
    g1[3] = (int)((unsigned)DDIM << 16);           // tile_dim0=256
    g1[4] = 16;                                    // tile_dim1=16
    g1[5] = DDIM;                                  // tensor_dim0_stride=256
    g1[6] = 0;
    g1[7] = 0;
    v4i_t z4 = {0, 0, 0, 0};
#if __clang_major__ >= 23
    v8i_t z8 = {0, 0, 0, 0, 0, 0, 0, 0};
    __builtin_amdgcn_tensor_load_to_lds(g0, g1, z4, z4, z8, 0);
#else
    __builtin_amdgcn_tensor_load_to_lds(g0, g1, z4, z4, 0);
#endif
}
#endif

// ---------------------------------------------------------------------------
// K0: zero scratch accumulators + loss output slot
// ---------------------------------------------------------------------------
__global__ void vq_init_kernel(float* __restrict__ hits, float* __restrict__ dw,
                               float* __restrict__ loss_slot) {
    int idx = blockIdx.x * 256 + threadIdx.x;
    if (idx < KCODES * DDIM) dw[idx] = 0.0f;
    if (idx < KCODES)        hits[idx] = 0.0f;
    if (idx == 0)            *loss_slot = 0.0f;
}

// ---------------------------------------------------------------------------
// K1: row-wise f32 -> bf16 conversion + squared L2 norm (used for x and w)
// blockDim must be 256 (== DDIM); one block per row.
// ---------------------------------------------------------------------------
__global__ __launch_bounds__(256) void vq_conv_rows_kernel(
    const float* __restrict__ src, unsigned short* __restrict__ dst,
    float* __restrict__ sq) {
    __shared__ float red[256];
    const int n = blockIdx.x;
    const int t = threadIdx.x;
    float v = src[(size_t)n * DDIM + t];
    dst[(size_t)n * DDIM + t] = f32_to_bf16_rne(v);
    red[t] = v * v;
    __syncthreads();
    for (int off = 128; off > 0; off >>= 1) {
        if (t < off) red[t] += red[t + off];
        __syncthreads();
    }
    if (t == 0) sq[n] = red[0];
}

// ---------------------------------------------------------------------------
// K2: distance GEMM (bf16 WMMA, f32 acc) fused with streaming Gumbel-max
// categorical sampling, hit counting and sparse dw scatter.
// Grid: NROWS/128 blocks x 256 threads (8 waves, 16 rows per wave).
// B tiles are TDM-staged into LDS (double buffered) when available.
// ---------------------------------------------------------------------------
__global__ __launch_bounds__(256) void vq_dist_sample_kernel(
    const float* __restrict__ x,
    const unsigned short* __restrict__ xb, const float* __restrict__ xsq,
    const unsigned short* __restrict__ wb, const float* __restrict__ wsq,
    int* __restrict__ samples, float* __restrict__ hits,
    float* __restrict__ dw) {
    __shared__ unsigned long long best_lds[128 * NSAMP];
#if VQ_USE_TDM
    __shared__ __align__(16) unsigned short bstage[2][16 * DDIM]; // 2 x 8 KiB
#endif

    const int tid = threadIdx.x;
    for (int i = tid; i < 128 * NSAMP; i += 256) best_lds[i] = 0ull;

    const int lane = tid & 31;
    const int wave = tid >> 5;
    const int m16  = lane & 15;   // row (A) / col (B) within 16-tile
    const int half = lane >> 4;   // lane half selects K/M sub-block
    const int rowbase = blockIdx.x * 128 + wave * 16;

#if VQ_USE_TDM
    // Prologue: wave 0 DMAs tile 0 into buffer 0, waits on TENSORcnt.
    if (wave == 0) {
        if (lane == 0)
            tdm_load_b_tile(wb, 0, (unsigned)(uintptr_t)&bstage[0][0]);
        __builtin_amdgcn_s_wait_tensorcnt(0);
    }
#endif
    __syncthreads();

    // Running argmax keys: rows (half*8 + r), 5 samples each.
    unsigned long long best[8][NSAMP];
#pragma unroll
    for (int r = 0; r < 8; ++r)
#pragma unroll
        for (int s = 0; s < NSAMP; ++s) best[r][s] = 0ull;

    // ||x||^2 for the 8 rows this lane's C-fragment covers.
    float xs[8];
#pragma unroll
    for (int r = 0; r < 8; ++r) xs[r] = xsq[rowbase + half * 8 + r];

    // Preload ALL A fragments (8 k-steps of 32) once; reused for all 64
    // column tiles of the codebook. A layout used here (16-bit 16x32, contiguous k, not the ISA table order):
    // lane: M = lane&15; VGPR j: K = (lane>>4)*16 + 2j (matches B fragment).
    FragBF16 afr[8];
#pragma unroll
    for (int ks = 0; ks < 8; ++ks) {
#pragma unroll
        for (int j = 0; j < 8; ++j) {
            const int kk = ks * 32 + half * 16 + j * 2;
            afr[ks].u[j] = *(const unsigned int*)(xb + (size_t)(rowbase + m16) * DDIM + kk);
        }
    }

    for (int ct = 0; ct < KCODES / 16; ++ct) {
        const int col = ct * 16 + m16;   // this lane's code column

#if VQ_USE_TDM
        // Prefetch next B tile into the other buffer while computing.
        if (wave == 0 && lane == 0 && (ct + 1) < KCODES / 16)
            tdm_load_b_tile(wb, ct + 1,
                            (unsigned)(uintptr_t)&bstage[(ct + 1) & 1][0]);
        const unsigned short* btile = &bstage[ct & 1][0];
#endif

        v8f acc = {0.f, 0.f, 0.f, 0.f, 0.f, 0.f, 0.f, 0.f};
#pragma unroll
        for (int ks = 0; ks < 8; ++ks) {
            // B layout used here (16-bit 32x16, contiguous k as in A): lane: N = lane&15;
            // VGPR j: K = (lane>>4)*16 + 2j  -> 8 contiguous dwords.
            FragBF16 bfr;
#if VQ_USE_TDM
            const uint4* bp =
                (const uint4*)(btile + m16 * DDIM + ks * 32 + half * 16);
            bfr.q[0] = bp[0];                 // ds_load_b128
            bfr.q[1] = bp[1];                 // ds_load_b128
#else
            const uint4* bp =
                (const uint4*)(wb + (size_t)col * DDIM + ks * 32 + half * 16);
            bfr.q[0] = bp[0];
            bfr.q[1] = bp[1];
#endif
            acc = __builtin_amdgcn_wmma_f32_16x16x32_bf16(
                false, afr[ks].v, false, bfr.v, (short)0, acc, false, false);
        }

        const float wsq_c = wsq[col];
#pragma unroll
        for (int r = 0; r < 8; ++r) {
            // C/D layout: acc[r] is (row = r + 8*half, col = lane&15)
            const float dist  = xs[r] + wsq_c - 2.0f * acc[r];
            const int   nglob = rowbase + half * 8 + r;
#pragma unroll
            for (int s = 0; s < NSAMP; ++s) {
                unsigned int h = mix32((unsigned int)nglob * 0x9E3779B9u
                                     ^ (unsigned int)col   * 0x85EBCA6Bu
                                     ^ (unsigned int)(s + 1) * 0xC2B2AE35u
                                     ^ 0x2545F491u /* seed(42)-derived */);
                // strictly in (0,1)
                const float u01 = ((float)(h >> 8) + 0.5f) * (1.0f / 16777216.0f);
                const float g   = -__logf(-__logf(u01));     // Gumbel(0,1)
                const float score = g - dist;                 // logits = -dist
                const unsigned long long key = pack_key(score, (unsigned int)col);
                if (key > best[r][s]) best[r][s] = key;
            }
        }

#if VQ_USE_TDM
        // Wait for the prefetched tile; barrier doubles as "done reading
        // buf[ct&1]" so the *next* prefetch may overwrite it.
        if (wave == 0)
            __builtin_amdgcn_s_wait_tensorcnt(0);
        __syncthreads();
#endif
    }

    // Cross-lane argmax via LDS 64-bit max atomics (ds_max_u64).
#pragma unroll
    for (int r = 0; r < 8; ++r) {
        const int lrow = wave * 16 + half * 8 + r;
#pragma unroll
        for (int s = 0; s < NSAMP; ++s)
            atomicMax(&best_lds[lrow * NSAMP + s], best[r][s]);
    }
    __syncthreads();

    // Emit samples + hit counts.
    for (int i = tid; i < 128 * NSAMP; i += 256) {
        const int lrow = i / NSAMP, s = i % NSAMP;
        const unsigned int k = (unsigned int)(best_lds[i] & 0xFFFFFFFFull);
        samples[(size_t)s * NROWS + blockIdx.x * 128 + lrow] = (int)k;
        atomicAdd(&hits[k], 1.0f);
    }
    __syncthreads();

    // Sparse dw scatter: dw[k, d] += x[n, d]/S  (thread tid owns dim d=tid).
    for (int lrow = 0; lrow < 128; ++lrow) {
        const int n = blockIdx.x * 128 + lrow;
        const float xv = x[(size_t)n * DDIM + tid] * (1.0f / NSAMP);
#pragma unroll
        for (int s = 0; s < NSAMP; ++s) {
            const unsigned int k =
                (unsigned int)(best_lds[lrow * NSAMP + s] & 0xFFFFFFFFull);
            atomicAdd(&dw[(size_t)k * DDIM + tid], xv);
        }
    }
}

// ---------------------------------------------------------------------------
// K3: EMA count normalization + perplexity. One block, 1024 threads (k = tid).
// ---------------------------------------------------------------------------
__global__ __launch_bounds__(1024) void vq_count_stats_kernel(
    const float* __restrict__ ema_count, const float* __restrict__ hits,
    float* __restrict__ out_count, float* __restrict__ out_ppl) {
    __shared__ float red[1024];
    const int k = threadIdx.x;
    const float hk = hits[k];
    const float c  = DECAYF * ema_count[k] + (1.0f - DECAYF) * (hk * (1.0f / NSAMP));
    red[k] = c;
    __syncthreads();
    for (int off = 512; off > 0; off >>= 1) {
        if (k < off) red[k] += red[k + off];
        __syncthreads();
    }
    const float nsum = red[0];
    __syncthreads();
    out_count[k] = (c + EPSF) / (nsum + (float)KCODES * EPSF) * nsum;

    const float p = hk / ((float)NSAMP * (float)NROWS);   // avg_probs[k]
    red[k] = p * __logf(p + 1e-10f);
    __syncthreads();
    for (int off = 512; off > 0; off >>= 1) {
        if (k < off) red[k] += red[k + off];
        __syncthreads();
    }
    if (k == 0) *out_ppl = __expf(-red[0]);
}

// ---------------------------------------------------------------------------
// K4: EMA weight + normalized embedding.
// ---------------------------------------------------------------------------
__global__ void vq_weight_kernel(const float* __restrict__ ema_weight,
                                 const float* __restrict__ dw,
                                 const float* __restrict__ new_count,
                                 float* __restrict__ out_weight,
                                 float* __restrict__ out_embed) {
    const int idx = blockIdx.x * 256 + threadIdx.x;
    if (idx >= KCODES * DDIM) return;
    const float nw = DECAYF * ema_weight[idx] + (1.0f - DECAYF) * dw[idx];
    out_weight[idx] = nw;
    out_embed[idx]  = nw / new_count[idx / DDIM];
}

// ---------------------------------------------------------------------------
// K5: gather with UPDATED codebook, straight-through output, commitment loss.
// One block per row, 256 threads (d = tid).
// ---------------------------------------------------------------------------
__global__ __launch_bounds__(256) void vq_gather_loss_kernel(
    const float* __restrict__ x, const float* __restrict__ new_embed,
    const int* __restrict__ samples, float* __restrict__ out_q,
    float* __restrict__ loss_out) {
    __shared__ float red[256];
    const int n = blockIdx.x;
    const int d = threadIdx.x;
    float q = 0.0f;
#pragma unroll
    for (int s = 0; s < NSAMP; ++s) {
        const int k = samples[(size_t)s * NROWS + n];
        q += new_embed[(size_t)k * DDIM + d];
    }
    q *= (1.0f / NSAMP);
    const float xv = x[(size_t)n * DDIM + d];
    out_q[(size_t)n * DDIM + d] = xv + (q - xv);   // straight-through
    const float diff = xv - q;
    red[d] = diff * diff;
    __syncthreads();
    for (int off = 128; off > 0; off >>= 1) {
        if (d < off) red[d] += red[d + off];
        __syncthreads();
    }
    if (d == 0)
        atomicAdd(loss_out, red[0] * (COMMITF / ((float)NROWS * (float)DDIM)));
}

// ---------------------------------------------------------------------------
// Host launcher
// ---------------------------------------------------------------------------
extern "C" void kernel_launch(void* const* d_in, const int* in_sizes, int n_in,
                              void* d_out, int out_size, void* d_ws, size_t ws_size,
                              hipStream_t stream) {
    (void)in_sizes; (void)n_in; (void)out_size; (void)ws_size;

    const float* x          = (const float*)d_in[0];  // [N, D]
    const float* embedding  = (const float*)d_in[1];  // [K, D]
    const float* ema_count  = (const float*)d_in[2];  // [K]
    const float* ema_weight = (const float*)d_in[3];  // [K, D]

    float* out = (float*)d_out;
    // Output layout (flat, return order): quantized_st, loss, perplexity,
    // new_embedding, new_count, new_weight.
    const size_t OFF_Q    = 0;
    const size_t OFF_LOSS = (size_t)NROWS * DDIM;                 // 8388608
    const size_t OFF_PPL  = OFF_LOSS + 1;
    const size_t OFF_EMB  = OFF_PPL + 1;
    const size_t OFF_CNT  = OFF_EMB + (size_t)KCODES * DDIM;
    const size_t OFF_W    = OFF_CNT + (size_t)KCODES;

    // Workspace layout (bytes)
    char* ws = (char*)d_ws;
    unsigned short* xb   = (unsigned short*)(ws + 0);                       // 16 MiB
    unsigned short* wb   = (unsigned short*)(ws + 16777216);                // 512 KiB
    float*          xsq  = (float*)(ws + 17301504);                         // 128 KiB
    float*          wsq  = (float*)(ws + 17432576);                         //   4 KiB
    float*          hits = (float*)(ws + 17436672);                         //   4 KiB
    float*          dw   = (float*)(ws + 17440768);                         //   1 MiB
    int*            smp  = (int*)(ws + 18489344);                           // 640 KiB

    // K0: zero accumulators (+ loss slot in d_out)
    vq_init_kernel<<<(KCODES * DDIM + 255) / 256, 256, 0, stream>>>(
        hits, dw, out + OFF_LOSS);

    // K1: bf16 conversion + row norms for x and w
    vq_conv_rows_kernel<<<NROWS, 256, 0, stream>>>(x, xb, xsq);
    vq_conv_rows_kernel<<<KCODES, 256, 0, stream>>>(embedding, wb, wsq);

    // K2: WMMA distance GEMM + fused Gumbel sampling + count/dw scatter
    vq_dist_sample_kernel<<<NROWS / 128, 256, 0, stream>>>(
        x, xb, xsq, wb, wsq, smp, hits, dw);

    // K3: EMA counts + perplexity
    vq_count_stats_kernel<<<1, 1024, 0, stream>>>(
        ema_count, hits, out + OFF_CNT, out + OFF_PPL);

    // K4: EMA weights + normalized embedding
    vq_weight_kernel<<<(KCODES * DDIM + 255) / 256, 256, 0, stream>>>(
        ema_weight, dw, out + OFF_CNT, out + OFF_W, out + OFF_EMB);

    // K5: gather with updated codebook + commitment loss
    vq_gather_loss_kernel<<<NROWS, 256, 0, stream>>>(
        x, out + OFF_EMB, smp, out + OFF_Q, out + OFF_LOSS);
}
